// SingleLayerLSTM_56289841381918
// MI455X (gfx1250) — matrix operand order repaired and checked by reading, not compiled, not run on hardware
//
#include <hip/hip_runtime.h>
#include <hip/hip_bf16.h>
#include <math.h>

// ---------------------------------------------------------------------------
// Single-layer LSTM for MI455X (gfx1250, wave32, WMMA).
//   B=128, T=1024, D=256, H=512 (4H=2048), O=256
// Strategy:
//   * persistent 64-WG scan kernel, device-wide barrier per timestep
//   * bf16x3 split GEMMs on v_wmma_f32_16x16x32_bf16 (fp32-class accuracy)
//   * W slices resident in LDS; h double-buffered in L2 as bf16 hi/lo
//   * x pre-split to bf16 hi/lo by a parallel pre-pass (off the critical path)
//   * x-projection of step t+1 computed in the barrier shadow (it does not
//     depend on h), so the serial chain is only: barrier -> h-GEMM -> cell
// ---------------------------------------------------------------------------

#define T_STEPS 1024
#define BATCH   128
#define DDIM    256
#define HDIM    512
#define GDIM    2048            // 4*H
#define ODIM    256
#define NWG     64              // persistent workgroups (must co-reside)
#define JPW     8               // hidden columns per WG  (HDIM / NWG)
#define NCOL    32              // gate columns per WG    (4 * JPW)
#define THREADS 256             // 8 waves of 32

typedef __bf16 bf16_t;
typedef __attribute__((ext_vector_type(16))) __bf16        v16bf;
typedef __attribute__((ext_vector_type(8)))  float         v8f;
typedef __attribute__((ext_vector_type(4)))  float         f32x4;
typedef __attribute__((ext_vector_type(4)))  unsigned int  u32x4;

// ---------------------------------------------------------------- helpers ---

__device__ __forceinline__ void bsplit(float x, bf16_t& hi, bf16_t& lo) {
  bf16_t h = (bf16_t)x;                 // RNE to bf16
  hi = h;
  lo = (bf16_t)(x - (float)h);          // residual captures next 8 mantissa bits
}

__device__ __forceinline__ v8f wmma_acc(v16bf a, v16bf b, v8f c) {
  // D = A(16x32) * B(32x16) + C(16x16 f32)
  return __builtin_amdgcn_wmma_f32_16x16x32_bf16(false, a, false, b,
                                                 (short)0, c, false, false);
}

// split product: (ah+al)(bh+bl) ~= ah*bh + al*bh + ah*bl
__device__ __forceinline__ v8f wmma_split(v16bf ah, v16bf al,
                                          v16bf bh, v16bf bl, v8f c) {
  c = wmma_acc(ah, bh, c);
  c = wmma_acc(al, bh, c);
  c = wmma_acc(ah, bl, c);
  return c;
}

// A fragment (16x32 bf16, row-major source, row length >= kb+32):
// lanes 0-15: K=[kb,kb+8) and [kb+16,kb+24); lanes 16-31: +8 each chunk.
__device__ __forceinline__ v16bf load_frag_a(const bf16_t* row, int lane, int kb) {
  const int c0 = kb + ((lane >> 4) & 1) * 8;
  union { v16bf v; u32x4 q[2]; } u;
  u.q[0] = *(const u32x4*)(row + c0);
  u.q[1] = *(const u32x4*)(row + c0 + 16);
  return u.v;
}

// B fragment (32x16 bf16). Source is W stored [n][k] row-major, so B(k,n)=W[n][k].
// lane L -> column n = L&15; lanes 0-15: K=[kb,kb+8) and [kb+16,kb+24); lanes 16-31: +8 each chunk.
__device__ __forceinline__ v16bf load_frag_b(const bf16_t* row, int lane, int kb) {
  const int c0 = kb + ((lane >> 4) & 1) * 8;
  union { v16bf v; u32x4 q[2]; } u;
  u.q[0] = *(const u32x4*)(row + c0);
  u.q[1] = *(const u32x4*)(row + c0 + 16);
  return u.v;
}

// A fragment from fp32 source with on-the-fly hi/lo bf16 split (fallback path).
__device__ __forceinline__ void load_frag_a_f32split(const float* row, int lane,
                                                     int kb, v16bf& hi, v16bf& lo) {
  const int c0 = kb + ((lane >> 4) & 1) * 8;
  f32x4 f0 = *(const f32x4*)(row + c0);
  f32x4 f1 = *(const f32x4*)(row + c0 + 4);
  f32x4 f2 = *(const f32x4*)(row + c0 + 16);
  f32x4 f3 = *(const f32x4*)(row + c0 + 20);
  float v4[16];
#pragma unroll
  for (int i = 0; i < 4; ++i) { v4[i] = f0[i]; v4[4+i] = f1[i]; v4[8+i] = f2[i]; v4[12+i] = f3[i]; }
  union { v16bf v; bf16_t e[16]; } uh, ul;
#pragma unroll
  for (int i = 0; i < 16; ++i) bsplit(v4[i], uh.e[i], ul.e[i]);
  hi = uh.v; lo = ul.v;
}

__device__ __forceinline__ float sigmoidf_dev(float z) {
  return 1.0f / (1.0f + __expf(-z));
}
__device__ __forceinline__ float tanhf_dev(float z) {
  // tanh(z) = 2*sigmoid(2z) - 1 ; saturates correctly for large |z|
  return 2.0f / (1.0f + __expf(-2.0f * z)) - 1.0f;
}

// ------------------------------------------------------------ prep kernels --

// Split weights to bf16 hi/lo, fold biases, zero h0 and the global barrier.
__global__ void __launch_bounds__(256)
lstm_prep_kernel(const float* __restrict__ W_ih, const float* __restrict__ W_hh,
                 const float* __restrict__ b_ih, const float* __restrict__ b_hh,
                 const float* __restrict__ W_out,
                 bf16_t* __restrict__ wih_hi, bf16_t* __restrict__ wih_lo,
                 bf16_t* __restrict__ whh_hi, bf16_t* __restrict__ whh_lo,
                 bf16_t* __restrict__ wout_hi, bf16_t* __restrict__ wout_lo,
                 float* __restrict__ bsum,
                 bf16_t* __restrict__ h0_hi, bf16_t* __restrict__ h0_lo,
                 unsigned* __restrict__ bar) {
  const int idx = blockIdx.x * 256 + threadIdx.x;
  if (idx < GDIM * HDIM) bsplit(W_hh[idx], whh_hi[idx], whh_lo[idx]);
  if (idx < GDIM * DDIM) bsplit(W_ih[idx], wih_hi[idx], wih_lo[idx]);
  if (idx < ODIM * HDIM) bsplit(W_out[idx], wout_hi[idx], wout_lo[idx]);
  if (idx < GDIM)        bsum[idx] = b_ih[idx] + b_hh[idx];
  if (idx < BATCH * HDIM) { h0_hi[idx] = (bf16_t)0.0f; h0_lo[idx] = (bf16_t)0.0f; }
  if (idx == 0)          *bar = 0u;
}

// Pre-split x (fp32 [B][T][D]) into bf16 hi/lo arrays, 4 elements per thread.
__global__ void __launch_bounds__(256)
lstm_xsplit_kernel(const float* __restrict__ x,
                   bf16_t* __restrict__ xh, bf16_t* __restrict__ xl) {
  const size_t i = ((size_t)blockIdx.x * 256 + threadIdx.x) * 4;
  f32x4 v = *(const f32x4*)(x + i);
  union { bf16_t e[4]; unsigned long long u; } ph, pl;
#pragma unroll
  for (int j = 0; j < 4; ++j) bsplit(v[j], ph.e[j], pl.e[j]);
  *(unsigned long long*)(xh + i) = ph.u;   // global_store_b64
  *(unsigned long long*)(xl + i) = pl.u;
}

// ------------------------------------------------------------ scan kernel ---
// Persistent: 64 WGs x 256 threads. WG `wg` owns hidden cols [wg*8, wg*8+8)
// => 32 gate columns (i,f,g,o x 8). Per step each wave computes a 16x32 tile
// of gates = x_t @ W_ih^T + h_{t-1} @ W_hh^T via split-bf16 WMMA. The x-term
// for step t+1 is computed between barrier-arrive and barrier-wait (it does
// not depend on h), hiding it in the barrier shadow.

template <bool XSPLIT>
__global__ void __launch_bounds__(THREADS, 1)
lstm_scan_kernel(const float* __restrict__ x,
                 const bf16_t* __restrict__ x_hi, const bf16_t* __restrict__ x_lo,
                 const bf16_t* __restrict__ wih_hi, const bf16_t* __restrict__ wih_lo,
                 const bf16_t* __restrict__ whh_hi, const bf16_t* __restrict__ whh_lo,
                 const float* __restrict__ bsum,
                 bf16_t* h_hi0, bf16_t* h_lo0, bf16_t* h_hi1, bf16_t* h_lo1,
                 unsigned* __restrict__ bar) {
  extern __shared__ char smem[];
  bf16_t* sWhhHi = (bf16_t*)smem;                 // [NCOL][HDIM]
  bf16_t* sWhhLo = sWhhHi + NCOL * HDIM;          // [NCOL][HDIM]
  bf16_t* sWihHi = sWhhLo + NCOL * HDIM;          // [NCOL][DDIM]
  bf16_t* sWihLo = sWihHi + NCOL * DDIM;          // [NCOL][DDIM]
  float*  sGates = (float*)(sWihLo + NCOL * DDIM);// [BATCH][NCOL]

  const int tid  = threadIdx.x;
  const int lane = tid & 31;
  const int wv   = tid >> 5;          // wave 0..7 -> M tile (batch rows 16*wv..)
  const int wg   = blockIdx.x;        // 0..63
  const int j0   = wg * JPW;

  // Stage this WG's weight slices into LDS (rows: n = gate_type*8 + jj).
  for (int idx = tid; idx < NCOL * HDIM; idx += THREADS) {
    int n = idx / HDIM, k = idx - n * HDIM;
    int r = (n >> 3) * HDIM + j0 + (n & 7);       // global gate row
    sWhhHi[n * HDIM + k] = whh_hi[(size_t)r * HDIM + k];
    sWhhLo[n * HDIM + k] = whh_lo[(size_t)r * HDIM + k];
  }
  for (int idx = tid; idx < NCOL * DDIM; idx += THREADS) {
    int n = idx / DDIM, k = idx - n * DDIM;
    int r = (n >> 3) * HDIM + j0 + (n & 7);
    sWihHi[n * DDIM + k] = wih_hi[(size_t)r * DDIM + k];
    sWihLo[n * DDIM + k] = wih_lo[(size_t)r * DDIM + k];
  }

  // Cell state + biases live in registers for the whole scan.
  float c_reg[4], Bi[4], Bf[4], Bg[4], Bo[4];
#pragma unroll
  for (int p = 0; p < 4; ++p) {
    c_reg[p] = 0.0f;
    int jj = (tid + THREADS * p) & 7;
    int jg = j0 + jj;
    Bi[p] = bsum[0 * HDIM + jg];
    Bf[p] = bsum[1 * HDIM + jg];
    Bg[p] = bsum[2 * HDIM + jg];
    Bo[p] = bsum[3 * HDIM + jg];
  }
  __syncthreads();

  const int arow = 16 * wv + (lane & 15);                 // batch row for A frags
  const int nlo  = lane & 15;                             // B-frag column lane
  const float*  xbase  = x    + (size_t)arow * T_STEPS * DDIM;
  const bf16_t* xhbase = XSPLIT ? x_hi + (size_t)arow * T_STEPS * DDIM : nullptr;
  const bf16_t* xlbase = XSPLIT ? x_lo + (size_t)arow * T_STEPS * DDIM : nullptr;

  // x-projection tile for timestep t (does NOT touch h -> overlappable).
  auto compute_xproj = [&](int t, v8f& a0, v8f& a1) {
    if (XSPLIT) {
      const bf16_t* xh = xhbase + (size_t)t * DDIM;
      const bf16_t* xl = xlbase + (size_t)t * DDIM;
      if (t + 1 < T_STEPS) {                       // prefetch the following step
        __builtin_prefetch(xh + DDIM, 0, 1);
        __builtin_prefetch(xl + DDIM, 0, 1);
      }
#pragma unroll
      for (int kb = 0; kb < DDIM; kb += 32) {
        v16bf ah = load_frag_a(xh, lane, kb);
        v16bf al = load_frag_a(xl, lane, kb);
        v16bf b0h = load_frag_b(sWihHi + (size_t)nlo * DDIM,        lane, kb);
        v16bf b0l = load_frag_b(sWihLo + (size_t)nlo * DDIM,        lane, kb);
        a0 = wmma_split(ah, al, b0h, b0l, a0);
        v16bf b1h = load_frag_b(sWihHi + (size_t)(16 + nlo) * DDIM, lane, kb);
        v16bf b1l = load_frag_b(sWihLo + (size_t)(16 + nlo) * DDIM, lane, kb);
        a1 = wmma_split(ah, al, b1h, b1l, a1);
      }
    } else {
      const float* xr = xbase + (size_t)t * DDIM;
      if (t + 1 < T_STEPS) __builtin_prefetch(xr + DDIM, 0, 1);
#pragma unroll 2
      for (int kb = 0; kb < DDIM; kb += 32) {
        v16bf ah, al;
        load_frag_a_f32split(xr, lane, kb, ah, al);
        v16bf b0h = load_frag_b(sWihHi + (size_t)nlo * DDIM,        lane, kb);
        v16bf b0l = load_frag_b(sWihLo + (size_t)nlo * DDIM,        lane, kb);
        a0 = wmma_split(ah, al, b0h, b0l, a0);
        v16bf b1h = load_frag_b(sWihHi + (size_t)(16 + nlo) * DDIM, lane, kb);
        v16bf b1l = load_frag_b(sWihLo + (size_t)(16 + nlo) * DDIM, lane, kb);
        a1 = wmma_split(ah, al, b1h, b1l, a1);
      }
    }
  };

  bf16_t* rdHi = h_hi0; bf16_t* rdLo = h_lo0;             // h_{t-1} (t=0: zeros)
  bf16_t* wrHi = h_hi1; bf16_t* wrLo = h_lo1;

  // Pipeline prologue: x-projection for t = 0.
  v8f xacc0 = {0.f,0.f,0.f,0.f,0.f,0.f,0.f,0.f};
  v8f xacc1 = {0.f,0.f,0.f,0.f,0.f,0.f,0.f,0.f};
  compute_xproj(0, xacc0, xacc1);

  for (int t = 0; t < T_STEPS; ++t) {
    v8f acc0 = xacc0;
    v8f acc1 = xacc1;

    // ---- recurrent projection slice: K = HDIM over h_{t-1} (critical path)
    const bf16_t* hrH = rdHi + (size_t)arow * HDIM;
    const bf16_t* hrL = rdLo + (size_t)arow * HDIM;
#pragma unroll
    for (int kb = 0; kb < HDIM; kb += 32) {
      v16bf ah = load_frag_a(hrH, lane, kb);
      v16bf al = load_frag_a(hrL, lane, kb);
      v16bf b0h = load_frag_b(sWhhHi + (size_t)nlo * HDIM,        lane, kb);
      v16bf b0l = load_frag_b(sWhhLo + (size_t)nlo * HDIM,        lane, kb);
      acc0 = wmma_split(ah, al, b0h, b0l, acc0);
      v16bf b1h = load_frag_b(sWhhHi + (size_t)(16 + nlo) * HDIM, lane, kb);
      v16bf b1l = load_frag_b(sWhhLo + (size_t)(16 + nlo) * HDIM, lane, kb);
      acc1 = wmma_split(ah, al, b1h, b1l, acc1);
    }

    // ---- spill gate tile to LDS (C layout: VGPR v, lane L -> M=v+8*(L>=16), N=L&15)
#pragma unroll
    for (int v = 0; v < 8; ++v) {
      int row = 16 * wv + v + ((lane >> 4) & 1) * 8;
      sGates[row * NCOL + nlo]      = acc0[v];
      sGates[row * NCOL + 16 + nlo] = acc1[v];
    }
    __syncthreads();

    // ---- pointwise LSTM cell: 256 threads x 4 (b, jj) pairs = 128*8 ----
#pragma unroll
    for (int p = 0; p < 4; ++p) {
      int lin = tid + THREADS * p;           // 0..1023
      int b = lin >> 3, jj = lin & 7;
      float gi = sGates[b * NCOL + jj]       + Bi[p];
      float gf = sGates[b * NCOL + 8  + jj]  + Bf[p];
      float gg = sGates[b * NCOL + 16 + jj]  + Bg[p];
      float go = sGates[b * NCOL + 24 + jj]  + Bo[p];
      float iv = sigmoidf_dev(gi);
      float fv = sigmoidf_dev(gf);
      float gv = tanhf_dev(gg);
      float ov = sigmoidf_dev(go);
      float cv = fv * c_reg[p] + iv * gv;
      c_reg[p] = cv;
      float hv = ov * tanhf_dev(cv);
      bf16_t hh, hl;
      bsplit(hv, hh, hl);
      wrHi[b * HDIM + j0 + jj] = hh;
      wrLo[b * HDIM + j0 + jj] = hl;
    }

    // ---- signal arrival (release), then fill the barrier shadow with the
    //      x-projection of step t+1 before waiting ----
    __threadfence();                 // release this thread's h stores
    __syncthreads();                 // all waves of the WG have released
    if (tid == 0)
      __hip_atomic_fetch_add(bar, 1u, __ATOMIC_RELEASE, __HIP_MEMORY_SCOPE_AGENT);

    xacc0 = (v8f){0.f,0.f,0.f,0.f,0.f,0.f,0.f,0.f};
    xacc1 = (v8f){0.f,0.f,0.f,0.f,0.f,0.f,0.f,0.f};
    if (t + 1 < T_STEPS) compute_xproj(t + 1, xacc0, xacc1);   // barrier shadow

    if (tid == 0) {
      const unsigned target = (unsigned)(t + 1) * NWG;
      while (__hip_atomic_load(bar, __ATOMIC_ACQUIRE, __HIP_MEMORY_SCOPE_AGENT) < target)
        __builtin_amdgcn_s_sleep(2);
    }
    __syncthreads();
    __threadfence();                 // acquire: invalidate stale L0 lines for new h

    // swap double buffers (write of step t becomes read of step t+1)
    bf16_t* th = rdHi; rdHi = wrHi; wrHi = th;
    bf16_t* tl = rdLo; rdLo = wrLo; wrLo = tl;
  }
  // T_STEPS even => final h landed in h_hi0/h_lo0.
}

// ---------------------------------------------------------- output kernel ---
// out[b][o] = sum_k h_T[b][k] * W_out[o][k] + b_out[o]  (split-bf16 WMMA)

__global__ void __launch_bounds__(256)
lstm_out_kernel(const bf16_t* __restrict__ h_hi, const bf16_t* __restrict__ h_lo,
                const bf16_t* __restrict__ wout_hi, const bf16_t* __restrict__ wout_lo,
                const float* __restrict__ b_out, float* __restrict__ out) {
  const int lane = threadIdx.x & 31;
  const int wid  = blockIdx.x * 8 + (threadIdx.x >> 5);  // 0..127
  const int mt = wid >> 4, nt = wid & 15;
  const int nlo = lane & 15;
  const bf16_t* ahr = h_hi    + (size_t)(mt * 16 + nlo) * HDIM;
  const bf16_t* alr = h_lo    + (size_t)(mt * 16 + nlo) * HDIM;
  const bf16_t* bhr = wout_hi + (size_t)(nt * 16 + nlo) * HDIM;
  const bf16_t* blr = wout_lo + (size_t)(nt * 16 + nlo) * HDIM;
  v8f acc = {0.f,0.f,0.f,0.f,0.f,0.f,0.f,0.f};
#pragma unroll 4
  for (int kb = 0; kb < HDIM; kb += 32) {
    v16bf ah = load_frag_a(ahr, lane, kb);
    v16bf al = load_frag_a(alr, lane, kb);
    v16bf bh = load_frag_b(bhr, lane, kb);
    v16bf bl = load_frag_b(blr, lane, kb);
    acc = wmma_split(ah, al, bh, bl, acc);
  }
#pragma unroll
  for (int v = 0; v < 8; ++v) {
    int row = mt * 16 + v + ((lane >> 4) & 1) * 8;
    int col = nt * 16 + nlo;
    out[row * ODIM + col] = acc[v] + b_out[col];
  }
}

// ------------------------------------------------------------------ launch ---

extern "C" void kernel_launch(void* const* d_in, const int* in_sizes, int n_in,
                              void* d_out, int out_size, void* d_ws, size_t ws_size,
                              hipStream_t stream) {
  const float* x     = (const float*)d_in[0];
  const float* W_ih  = (const float*)d_in[1];
  const float* W_hh  = (const float*)d_in[2];
  const float* b_ih  = (const float*)d_in[3];
  const float* b_hh  = (const float*)d_in[4];
  const float* W_out = (const float*)d_in[5];
  const float* b_out = (const float*)d_in[6];
  float* out = (float*)d_out;

  // -------- workspace carve-up (base ~7.2 MB; +134 MB for x hi/lo) --------
  char* ws = (char*)d_ws;
  size_t off = 0;
  auto take = [&](size_t bytes) -> char* {
    char* p = ws + off;
    off += (bytes + 255) & ~(size_t)255;
    return p;
  };
  unsigned* bar   = (unsigned*)take(sizeof(unsigned));
  bf16_t* h_hi0   = (bf16_t*)take((size_t)BATCH * HDIM * sizeof(bf16_t));
  bf16_t* h_lo0   = (bf16_t*)take((size_t)BATCH * HDIM * sizeof(bf16_t));
  bf16_t* h_hi1   = (bf16_t*)take((size_t)BATCH * HDIM * sizeof(bf16_t));
  bf16_t* h_lo1   = (bf16_t*)take((size_t)BATCH * HDIM * sizeof(bf16_t));
  bf16_t* wih_hi  = (bf16_t*)take((size_t)GDIM * DDIM * sizeof(bf16_t));
  bf16_t* wih_lo  = (bf16_t*)take((size_t)GDIM * DDIM * sizeof(bf16_t));
  bf16_t* whh_hi  = (bf16_t*)take((size_t)GDIM * HDIM * sizeof(bf16_t));
  bf16_t* whh_lo  = (bf16_t*)take((size_t)GDIM * HDIM * sizeof(bf16_t));
  bf16_t* wout_hi = (bf16_t*)take((size_t)ODIM * HDIM * sizeof(bf16_t));
  bf16_t* wout_lo = (bf16_t*)take((size_t)ODIM * HDIM * sizeof(bf16_t));
  float*  bsum    = (float*)take((size_t)GDIM * sizeof(float));

  const size_t xelems = (size_t)BATCH * T_STEPS * DDIM;   // 33,554,432
  bf16_t* x_hi = (bf16_t*)take(xelems * sizeof(bf16_t));
  bf16_t* x_lo = (bf16_t*)take(xelems * sizeof(bf16_t));
  const bool use_xsplit = (off <= ws_size);               // deterministic per-size
  (void)in_sizes; (void)n_in; (void)out_size;

  // 1) split weights to bf16 hi/lo, fold biases, zero h0 + barrier
  {
    int n = GDIM * HDIM;                 // largest range (W_hh)
    lstm_prep_kernel<<<(n + 255) / 256, 256, 0, stream>>>(
        W_ih, W_hh, b_ih, b_hh, W_out,
        wih_hi, wih_lo, whh_hi, whh_lo, wout_hi, wout_lo,
        bsum, h_hi0, h_lo0, bar);
  }

  // 1b) pre-split x into bf16 hi/lo (parallel, off the recurrent critical path)
  if (use_xsplit) {
    unsigned nblk = (unsigned)(xelems / (256 * 4));       // 4 elems/thread
    lstm_xsplit_kernel<<<nblk, 256, 0, stream>>>(x, x_hi, x_lo);
  }

  // 2) persistent recurrent scan (64 co-resident WGs, grid barrier per step)
  {
    size_t smem = (size_t)(2 * NCOL * HDIM + 2 * NCOL * DDIM) * sizeof(bf16_t)
                + (size_t)BATCH * NCOL * sizeof(float);   // 114688 B < 320 KB/WGP
    if (use_xsplit) {
      lstm_scan_kernel<true><<<NWG, THREADS, smem, stream>>>(
          x, x_hi, x_lo, wih_hi, wih_lo, whh_hi, whh_lo, bsum,
          h_hi0, h_lo0, h_hi1, h_lo1, bar);
    } else {
      lstm_scan_kernel<false><<<NWG, THREADS, smem, stream>>>(
          x, x_hi, x_lo, wih_hi, wih_lo, whh_hi, whh_lo, bsum,
          h_hi0, h_lo0, h_hi1, h_lo1, bar);
    }
  }

  // 3) output projection: out = h_T @ W_out^T + b_out
  lstm_out_kernel<<<16, 256, 0, stream>>>(h_hi0, h_lo0, wout_hi, wout_lo, b_out, out);
}
